// ScaledDotProductAttention_77257871721148
// MI455X (gfx1250) — matrix operand order repaired and checked by reading, not compiled, not run on hardware
//
#include <hip/hip_runtime.h>

// ---------------------------------------------------------------------------
// Causal attention, fp32 in/out, f16 WMMA core (gfx1250 / CDNA5, wave32).
//   d_out = [ output : B*H*S*D floats | attn : B*H*S*S floats ]
// Pre-pass (if ws fits): K -> f16*(1/8) row-major, V -> f16 transposed [d][s];
// WMMA operands then load as contiguous b128 pairs with zero conversion VALU.
// One wave owns one 16-row query tile; two sweeps over its causal key range:
//   sweep 1: online (m, l) via v_wmma_f32_16x16x32_f16 + lane-shuffle reduce
//   sweep 2: recompute scores, write attn = exp(s-m)/l, accumulate O = P@V
// ---------------------------------------------------------------------------

typedef __attribute__((ext_vector_type(16))) _Float16 v16h;
typedef __attribute__((ext_vector_type(4)))  _Float16 v4h;
typedef __attribute__((ext_vector_type(8)))  float    v8f;
typedef __attribute__((ext_vector_type(4)))  float    v4f;

#define BATCH 4
#define HEADS 16
#define SEQ   2048
#define DIM   64
#define L2E   1.44269504088896340736f

__device__ __forceinline__ v8f wmma_f16(v16h a, v16h b, v8f c) {
  return __builtin_amdgcn_wmma_f32_16x16x32_f16(false, a, false, b,
                                                (short)0, c, false, false);
}

// reductions across the 16 lanes of each wave half (WMMA C-layout columns)
__device__ __forceinline__ float rowmax16(float v) {
  v = fmaxf(v, __shfl_xor(v, 1, 32));
  v = fmaxf(v, __shfl_xor(v, 2, 32));
  v = fmaxf(v, __shfl_xor(v, 4, 32));
  v = fmaxf(v, __shfl_xor(v, 8, 32));
  return v;
}
__device__ __forceinline__ float rowsum16(float v) {
  v += __shfl_xor(v, 1, 32);
  v += __shfl_xor(v, 2, 32);
  v += __shfl_xor(v, 4, 32);
  v += __shfl_xor(v, 8, 32);
  return v;
}

// A operand (16x32 f16): lane<16 row=lane K={0..15}; lane>=16 K={16..31}
__device__ __forceinline__ v16h load_qa(const float* __restrict__ qb, int q0,
                                        int slice, int lh, int hi) {
  const float* qr = qb + (size_t)(q0 + lh) * DIM + slice * 32 + hi * 16;
  v16h a;
#pragma unroll
  for (int j = 0; j < 8; ++j) a[j] = (_Float16)qr[j];
#pragma unroll
  for (int j = 0; j < 8; ++j) a[8 + j] = (_Float16)qr[8 + j];
  return a;
}

// B operand (32x16 f16) for K^T: lane = key column; halves = contraction dims.
// F16 path: 16 contiguous halfs -> two global_load_b128, no VALU.
template <bool F16>
__device__ __forceinline__ v16h load_kb(const float* __restrict__ kb,
                                        const _Float16* __restrict__ kf,
                                        int kt, int slice, int lh, int hi) {
  if constexpr (F16) {
    const _Float16* kr = kf + (size_t)(kt + lh) * DIM + slice * 32 + hi * 16;
    return *(const v16h*)kr;
  } else {
    const float* kr = kb + (size_t)(kt + lh) * DIM + slice * 32 + hi * 16;
    v16h b;
#pragma unroll
    for (int j = 0; j < 16; ++j) b[j] = (_Float16)(kr[j] * 0.125f);
    return b;
  }
}

template <bool F16>
__device__ __forceinline__ void score_tiles(const float* __restrict__ Kb,
                                            const _Float16* __restrict__ Kf,
                                            const v16h& aq0, const v16h& aq1,
                                            int k0, int q0, int lh, int hi,
                                            v8f& s0, v8f& s1) {
  v8f z0 = {}, z1 = {};
  z0 = wmma_f16(aq0, load_kb<F16>(Kb, Kf, k0, 0, lh, hi), z0);
  z0 = wmma_f16(aq1, load_kb<F16>(Kb, Kf, k0, 1, lh, hi), z0);
  z1 = wmma_f16(aq0, load_kb<F16>(Kb, Kf, k0 + 16, 0, lh, hi), z1);
  z1 = wmma_f16(aq1, load_kb<F16>(Kb, Kf, k0 + 16, 1, lh, hi), z1);
  if (k0 + 31 > q0) {                    // group straddles/crosses the diagonal
    const int key0 = k0 + lh, key1 = k0 + 16 + lh;
#pragma unroll
    for (int r = 0; r < 8; ++r) {
      const int row = q0 + r + hi * 8;
      z0[r] = (key0 > row) ? -1.0e9f : z0[r];
      z1[r] = (key1 > row) ? -1.0e9f : z1[r];
    }
  }
  s0 = z0; s1 = z1;
}

template <bool F16>
__global__ __launch_bounds__(128) void
ScaledDotProductAttention_77257871721148_kernel(const float* __restrict__ Q,
                                                const float* __restrict__ K,
                                                const float* __restrict__ V,
                                                const _Float16* __restrict__ Kf,
                                                const _Float16* __restrict__ Vt,
                                                float* __restrict__ Out,
                                                float* __restrict__ Attn) {
  // per-wave P staging tile: 16 rows x 32 cols f16, stride 40 halfs (80 B)
  // -> 16B-aligned A-layout reads, bank-conflict-free across the 16 lanes.
  __shared__ _Float16 lds_p[4][16 * 40];

  const int lane = threadIdx.x & 31;
  const int wave = threadIdx.x >> 5;
  const int lh   = lane & 15;
  const int hi   = (lane >> 4) & 1;

  const int bh = blockIdx.x >> 5;            // head index 0..63
  const int qb = blockIdx.x & 31;            // 64-row block within sequence
  const int q0 = qb * 64 + wave * 16;        // this wave's query tile

  const float*    Qb  = Q + (size_t)bh * SEQ * DIM;
  const float*    Kb  = K + (size_t)bh * SEQ * DIM;
  const float*    Vb  = V + (size_t)bh * SEQ * DIM;
  const _Float16* Kfb = Kf + (size_t)bh * SEQ * DIM;   // [s][d] f16, pre-scaled
  const _Float16* Vtb = Vt + (size_t)bh * DIM * SEQ;   // [d][s] f16 (transposed)
  float*          Ob  = Out + (size_t)bh * SEQ * DIM;
  float*          Ab  = Attn + (size_t)bh * SEQ * SEQ;

  const v16h aq0 = load_qa(Qb, q0, 0, lh, hi);   // Q dims 0..31
  const v16h aq1 = load_qa(Qb, q0, 1, lh, hi);   // Q dims 32..63

  const int ngroups = (q0 + 16 + 31) >> 5;       // 32-key groups in causal span

  float m[8], l[8];
#pragma unroll
  for (int r = 0; r < 8; ++r) { m[r] = -3.0e38f; l[r] = 0.0f; }

  // ----------------- Sweep 1: online softmax statistics (m, l) -------------
  for (int g = 0; g < ngroups; ++g) {
    const int k0 = g << 5;
    if (g + 1 < ngroups) {
      if constexpr (F16)
        __builtin_prefetch(Kfb + (size_t)(k0 + 32 + lh) * DIM, 0, 3);
      else
        __builtin_prefetch(Kb + (size_t)(k0 + 32 + lh) * DIM, 0, 3);
    }
    v8f s0, s1;
    score_tiles<F16>(Kb, Kfb, aq0, aq1, k0, q0, lh, hi, s0, s1);
#pragma unroll
    for (int r = 0; r < 8; ++r) {
      float t  = rowmax16(fmaxf(s0[r], s1[r]));
      float mn = fmaxf(m[r], t);
      float alpha = __builtin_amdgcn_exp2f((m[r] - mn) * L2E);
      float e0 = __builtin_amdgcn_exp2f((s0[r] - mn) * L2E);
      float e1 = __builtin_amdgcn_exp2f((s1[r] - mn) * L2E);
      float rs = rowsum16(e0 + e1);
      l[r] = l[r] * alpha + rs;
      m[r] = mn;
    }
  }
  float rl[8];
#pragma unroll
  for (int r = 0; r < 8; ++r) rl[r] = __builtin_amdgcn_rcpf(l[r]);

  // ----------------- Sweep 2: write attn, accumulate O = P @ V -------------
  v8f o0 = {}, o1 = {}, o2 = {}, o3 = {};
  _Float16* lp = lds_p[wave];

  for (int g = 0; g < ngroups; ++g) {
    const int k0 = g << 5;
    if (g + 1 < ngroups) {
      if constexpr (F16) {
        __builtin_prefetch(Kfb + (size_t)(k0 + 32 + lh) * DIM, 0, 3);
        __builtin_prefetch(Vtb + (size_t)(lh * 4) * SEQ + k0 + 32, 0, 3);
      } else {
        __builtin_prefetch(Kb + (size_t)(k0 + 32 + lh) * DIM, 0, 3);
        __builtin_prefetch(Vb + (size_t)(k0 + 32 + lh) * DIM, 0, 3);
      }
    }
    v8f s0, s1;
    score_tiles<F16>(Kb, Kfb, aq0, aq1, k0, q0, lh, hi, s0, s1);

    // p = exp(s - m) / l ; store to global attn and to LDS (f16, C-layout)
#pragma unroll
    for (int r = 0; r < 8; ++r) {
      const int row = q0 + r + hi * 8;
      float p0 = __builtin_amdgcn_exp2f((s0[r] - m[r]) * L2E) * rl[r];
      float p1 = __builtin_amdgcn_exp2f((s1[r] - m[r]) * L2E) * rl[r];
      float* arow = Ab + (size_t)row * SEQ;
      arow[k0 + lh]      = p0;
      arow[k0 + 16 + lh] = p1;
      lp[(r + hi * 8) * 40 + lh]      = (_Float16)p0;
      lp[(r + hi * 8) * 40 + 16 + lh] = (_Float16)p1;
    }

    // re-read P in WMMA A-layout (transpose via LDS; same-wave DS ordering)
    v16h pa;
    {
      const _Float16* prow = lp + lh * 40 + hi * 16;
#pragma unroll
      for (int j = 0; j < 8; ++j) pa[j] = prow[j];          // K 0..7 / 16..23
#pragma unroll
      for (int j = 0; j < 8; ++j) pa[8 + j] = prow[8 + j];  // K 8..15 / 24..31
    }

    // O[:, 16a..16a+15] += P(16x32) @ V(32x16)
#pragma unroll
    for (int a = 0; a < 4; ++a) {
      v16h bv;
      if constexpr (F16) {
        // Vt[d][s]: lane = output dim, 16 contiguous keys -> two b128 loads
        const _Float16* vr = Vtb + (size_t)(a * 16 + lh) * SEQ + k0 + hi * 16;
        bv = *(const v16h*)vr;
      } else {
        const float* vc = Vb + (size_t)(k0 + hi * 16) * DIM + a * 16 + lh;
#pragma unroll
        for (int j = 0; j < 16; ++j) bv[j] = (_Float16)vc[j * DIM];
      }
      if      (a == 0) o0 = wmma_f16(pa, bv, o0);
      else if (a == 1) o1 = wmma_f16(pa, bv, o1);
      else if (a == 2) o2 = wmma_f16(pa, bv, o2);
      else             o3 = wmma_f16(pa, bv, o3);
    }
  }

  // ----------------- Store O (already normalized by 1/l) -------------------
#pragma unroll
  for (int r = 0; r < 8; ++r) {
    float* orow = Ob + (size_t)(q0 + r + hi * 8) * DIM + lh;
    orow[0]  = o0[r];
    orow[16] = o1[r];
    orow[32] = o2[r];
    orow[48] = o3[r];
  }

  // ----------------- Zero-fill the strictly-masked attn columns -----------
  const int covered = ngroups << 5;          // columns written by sweep 2
  const v4f z4 = {0.0f, 0.0f, 0.0f, 0.0f};
  for (int r = 0; r < 16; ++r) {
    float* arow = Ab + (size_t)(q0 + r) * SEQ;
    for (int c = covered + lane * 4; c < SEQ; c += 128)
      *(v4f*)(arow + c) = z4;
  }
}

// --------------------- pre-pass: K -> f16 * (1/sqrt(D)) --------------------
__global__ __launch_bounds__(256) void k_to_f16_kernel(const float* __restrict__ K,
                                                       _Float16* __restrict__ Kf) {
  const size_t i = ((size_t)blockIdx.x * 256 + threadIdx.x) * 4;
  const v4f x = *(const v4f*)(K + i);
  v4h y;
#pragma unroll
  for (int j = 0; j < 4; ++j) y[j] = (_Float16)(x[j] * 0.125f);
  *(v4h*)(Kf + i) = y;
}

// --------------------- pre-pass: V -> f16, transposed [bh][d][s] -----------
__global__ __launch_bounds__(256) void v_tr_f16_kernel(const float* __restrict__ V,
                                                       _Float16* __restrict__ Vt) {
  __shared__ _Float16 tile[32][33];
  const int tx = threadIdx.x & 31;
  const int ty = threadIdx.x >> 5;            // 0..7
  const int bh = blockIdx.z;
  const int k0 = blockIdx.x * 32;
  const int d0 = blockIdx.y * 32;
  const float* vb = V + (size_t)bh * SEQ * DIM;
#pragma unroll
  for (int j = 0; j < 4; ++j) {
    const int k = ty + 8 * j;
    tile[k][tx] = (_Float16)vb[(size_t)(k0 + k) * DIM + d0 + tx];
  }
  __syncthreads();
  _Float16* vt = Vt + (size_t)bh * DIM * SEQ;
#pragma unroll
  for (int j = 0; j < 4; ++j) {
    const int d = ty + 8 * j;
    vt[(size_t)(d0 + d) * SEQ + k0 + tx] = tile[tx][d];
  }
}

extern "C" void kernel_launch(void* const* d_in, const int* in_sizes, int n_in,
                              void* d_out, int out_size, void* d_ws, size_t ws_size,
                              hipStream_t stream) {
  const float* Q = (const float*)d_in[0];
  const float* K = (const float*)d_in[1];
  const float* V = (const float*)d_in[2];
  // d_in[3] = attn_mask (bool causal triu, k=1) -- structure is hard-coded.

  float* out  = (float*)d_out;                                  // [B,H,S,D]
  float* attn = out + (size_t)BATCH * HEADS * SEQ * DIM;        // [B,H,S,S]

  const size_t nkv = (size_t)BATCH * HEADS * SEQ * DIM;         // 8,388,608
  const bool f16path = ws_size >= nkv * 2 * sizeof(_Float16);   // 33.5 MB

  const dim3 grid(BATCH * HEADS * (SEQ / 64));   // 2048 blocks
  const dim3 block(128);                         // 4 waves; one q-tile each

  if (f16path) {
    _Float16* Kf = (_Float16*)d_ws;
    _Float16* Vt = Kf + nkv;
    k_to_f16_kernel<<<(unsigned)(nkv / (256 * 4)), 256, 0, stream>>>(K, Kf);
    v_tr_f16_kernel<<<dim3(SEQ / 32, DIM / 32, BATCH * HEADS), 256, 0, stream>>>(V, Vt);
    ScaledDotProductAttention_77257871721148_kernel<true>
        <<<grid, block, 0, stream>>>(Q, K, V, Kf, Vt, out, attn);
  } else {
    ScaledDotProductAttention_77257871721148_kernel<false>
        <<<grid, block, 0, stream>>>(Q, K, V, nullptr, nullptr, out, attn);
  }
}
